// MultiHeadAttention_53429393162746
// MI455X (gfx1250) — matrix operand order repaired and checked by reading, not compiled, not run on hardware
//
#include <hip/hip_runtime.h>
#include <hip/hip_bf16.h>

typedef __bf16 bf16;
typedef __attribute__((ext_vector_type(16))) __bf16 v16bf;
typedef __attribute__((ext_vector_type(8)))  __bf16 v8bf;
typedef __attribute__((ext_vector_type(8)))  float  v8f;
typedef __attribute__((ext_vector_type(4)))  unsigned int v4u;
typedef __attribute__((ext_vector_type(8)))  int v8i;
typedef __attribute__((ext_vector_type(4)))  int v4i;

#define DEV static __device__ __forceinline__

#if __has_builtin(__builtin_amdgcn_tensor_load_to_lds)
#define USE_TDM 1
#else
#define USE_TDM 0
#endif

// ---------------------------------------------------------------------------
// WMMA fragment helpers (bf16 16x16x32, wave32 layouts per CDNA5 ISA 7.12.2)
// ---------------------------------------------------------------------------
DEV v16bf load_A_frag(const bf16* base, int ld) {
  const int lane = threadIdx.x & 31;
  const int row  = lane & 15;
  const int ko   = (lane >> 4) * 8;
  const bf16* p = base + row * ld + ko;
  union { v16bf v; v8bf h[2]; } u;
  u.h[0] = *(const v8bf*)(p);
  u.h[1] = *(const v8bf*)(p + 16);
  return u.v;
}

DEV v16bf load_B_frag(const bf16* base, int ld) {
  const int lane = threadIdx.x & 31;
  const int n    = lane & 15;
  const int ko   = (lane >> 4) * 8;
  const bf16* p = base + n * ld + ko;
  union { v16bf v; v8bf h[2]; } u;
  u.h[0] = *(const v8bf*)(p);
  u.h[1] = *(const v8bf*)(p + 16);
  return u.v;
}

DEV v8f wmma_bf16(v16bf a, v16bf b, v8f c) {
  return __builtin_amdgcn_wmma_f32_16x16x32_bf16(false, a, false, b, (short)0, c,
                                                 false, false);
}

// ---------------------------------------------------------------------------
// TDM 2D tile load: global [tile_d1 rows x tile_d0 elems] (bf16, row stride
// stride0 elems) -> LDS at byte offset lds_off, with optional DWORD padding
// after every row (pad codes per CDNA5 ISA 8.4).  D# built per ISA 8.3/8.4.
// This toolchain exposes the 6-arg builtin:
//   (v4u g0, v8i g1, v4i g2, v4i g3, v8i extra, i32 cpol)
// ---------------------------------------------------------------------------
DEV void tdm_load_2d(unsigned lds_off, const void* gaddr,
                     unsigned tile_d0, unsigned tile_d1,
                     unsigned long long stride0,
                     unsigned pad_interval_code, unsigned pad_amount_code,
                     bool pad_en) {
#if USE_TDM
  unsigned long long ga = (unsigned long long)(size_t)gaddr;
  v4u g0;
  g0[0] = 1u;                                       // count=1, user mode
  g0[1] = lds_off;                                  // lds_addr (bytes)
  g0[2] = (unsigned)(ga & 0xffffffffu);             // global_addr[31:0]
  g0[3] = (unsigned)((ga >> 32) & 0x01ffffffu) | (2u << 30);  // type=2 (image)
  v8i g1;
  g1[0] = (int)((1u << 16) |                        // data_size = 2 bytes
                (pad_en ? (1u << 20) : 0u) |        // pad_enable
                (pad_interval_code << 22) | (pad_amount_code << 25));
  g1[1] = (int)((tile_d0 & 0xffffu) << 16);         // tensor_dim0[15:0]
  g1[2] = (int)((tile_d0 >> 16) | ((tile_d1 & 0xffffu) << 16)); // dim0 hi|dim1 lo
  g1[3] = (int)((tile_d1 >> 16) | (tile_d0 << 16)); // dim1 hi | tile_dim0
  g1[4] = (int)(tile_d1 & 0xffffu);                 // tile_dim1 (tile_dim2=0)
  g1[5] = (int)(stride0 & 0xffffffffull);           // tensor_dim0_stride lo
  g1[6] = (int)((stride0 >> 32) & 0xffffull);       // stride hi (dim1_stride=0)
  g1[7] = 0;
  v4i g2 = {0, 0, 0, 0};
  v4i g3 = {0, 0, 0, 0};
  v8i gx = {0, 0, 0, 0, 0, 0, 0, 0};
  __builtin_amdgcn_tensor_load_to_lds(g0, g1, g2, g3, gx, 0);
#else
  (void)lds_off; (void)gaddr; (void)tile_d0; (void)tile_d1; (void)stride0;
  (void)pad_interval_code; (void)pad_amount_code; (void)pad_en;
#endif
}

// ---------------------------------------------------------------------------
// Kernel 1a: fp32 -> bf16 elementwise
// ---------------------------------------------------------------------------
__global__ void k_f32_to_bf16(const float* __restrict__ in, bf16* __restrict__ out,
                              int n) {
  int i = blockIdx.x * blockDim.x + threadIdx.x;
  if (i < n) out[i] = (bf16)in[i];
}

// ---------------------------------------------------------------------------
// Kernel 1b: transpose-convert [H][R][C] fp32 -> [H][C][R] bf16
// ---------------------------------------------------------------------------
__global__ void k_transpose_w(const float* __restrict__ in, bf16* __restrict__ out,
                              int H, int R, int C) {
  int i = blockIdx.x * blockDim.x + threadIdx.x;
  int n = H * R * C;
  if (i >= n) return;
  int h = i / (R * C);
  int rem = i - h * (R * C);
  int r = rem / C;
  int c = rem - r * C;
  out[(size_t)h * R * C + (size_t)c * R + r] = (bf16)in[i];
}

// ---------------------------------------------------------------------------
// Kernel 2: batched QKV projection GEMM.  WG tile 128x64, 4 waves; wave w owns
// rows w*32..w*32+31 (2 A frags) x 64 cols (4 B frags) -> 8 WMMA / K-step.
// z = h*3 + proj.  Q,K row-major [h][8192][512]; V transposed [h][4][512][2048].
// ---------------------------------------------------------------------------
__global__ __launch_bounds__(128) void k_qkv_gemm(
    const bf16* __restrict__ xb,
    const bf16* __restrict__ wqT, const bf16* __restrict__ wkT,
    const bf16* __restrict__ wvT,
    const float* __restrict__ bq, const float* __restrict__ bk,
    const float* __restrict__ bv,
    bf16* __restrict__ Qb, bf16* __restrict__ Kb, bf16* __restrict__ Vt) {
  const int D = 512;
  __shared__ __align__(16) bf16 As[128 * 40];
  __shared__ __align__(16) bf16 Bs[64 * 40];
  const int m0 = blockIdx.x * 128;
  const int n0 = blockIdx.y * 64;
  const int z  = blockIdx.z;
  const int h = z / 3, proj = z % 3;
  const bf16* WT = (proj == 0 ? wqT : proj == 1 ? wkT : wvT) + (size_t)h * D * D;
  const float* bias = (proj == 0 ? bq : proj == 1 ? bk : bv) + h * D;
  const int tid = threadIdx.x, w = tid >> 5, lane = tid & 31;

  v8f acc[2][4];
#pragma unroll
  for (int mi = 0; mi < 2; ++mi)
#pragma unroll
    for (int t = 0; t < 4; ++t)
#pragma unroll
      for (int j = 0; j < 8; ++j) acc[mi][t][j] = 0.0f;

  for (int kk = 0; kk < D; kk += 32) {
#pragma unroll
    for (int it = 0; it < 4; ++it) {
      int c = tid + it * 128;
      int r = c >> 2, ko = (c & 3) * 8;
      *(v8bf*)&As[r * 40 + ko] = *(const v8bf*)&xb[(size_t)(m0 + r) * D + kk + ko];
    }
#pragma unroll
    for (int it = 0; it < 2; ++it) {
      int c = tid + it * 128;
      int r = c >> 2, ko = (c & 3) * 8;
      *(v8bf*)&Bs[r * 40 + ko] = *(const v8bf*)&WT[(size_t)(n0 + r) * D + kk + ko];
    }
    __syncthreads();
    v16bf a0 = load_A_frag(&As[(w * 32) * 40], 40);
    v16bf a1 = load_A_frag(&As[(w * 32 + 16) * 40], 40);
#pragma unroll
    for (int t = 0; t < 4; ++t) {
      v16bf b = load_B_frag(&Bs[t * 16 * 40], 40);
      acc[0][t] = wmma_bf16(a0, b, acc[0][t]);
      acc[1][t] = wmma_bf16(a1, b, acc[1][t]);
    }
    __syncthreads();
  }

#pragma unroll
  for (int mi = 0; mi < 2; ++mi)
#pragma unroll
    for (int t = 0; t < 4; ++t) {
      int col = n0 + t * 16 + (lane & 15);
      float bval = bias[col];
#pragma unroll
      for (int j = 0; j < 8; ++j) {
        int row = m0 + w * 32 + mi * 16 + j + (lane >> 4) * 8;
        float vout = acc[mi][t][j] + bval;
        if (proj == 2) {
          int bb = row >> 11, ss = row & 2047;
          Vt[(((size_t)h * 4 + bb) * 512 + col) * 2048 + ss] = (bf16)vout;
        } else {
          bf16* dst = (proj == 0 ? Qb : Kb);
          dst[((size_t)h * 8192 + row) * 512 + col] = (bf16)vout;
        }
      }
    }
}

// ---------------------------------------------------------------------------
// Kernel 3: causal flash attention.
//   Grid (S/64, B, H), 128 threads = 4 waves. Wave w: 16 queries, full d=512.
//   LDS: Q 64x520, K 32x520, Vt 512x40, P 4x16x40 (padded strides).
//   K/V/Q tiles staged by the Tensor Data Mover when available.
// ---------------------------------------------------------------------------
#define QS_OFF 0u
#define KS_OFF (64u * 520u * 2u)                 // 66560
#define VS_OFF (KS_OFF + 32u * 520u * 2u)        // 99840
#define PS_OFF (VS_OFF + 512u * 40u * 2u)        // 140800

__global__ __launch_bounds__(128) void k_flash(
    const bf16* __restrict__ Qb, const bf16* __restrict__ Kb,
    const bf16* __restrict__ Vt, bf16* __restrict__ cat) {
  const int D = 512, S = 2048;
  extern __shared__ char smem_raw[];
  bf16* qs = (bf16*)smem_raw;          // 64 x 520
  bf16* ks = qs + 64 * 520;            // 32 x 520
  bf16* vs = ks + 32 * 520;            // 512 x 40
  bf16* ps = vs + 512 * 40;            // 4 x 16 x 40

  const int q0 = blockIdx.x * 64;
  const int b  = blockIdx.y;
  const int h  = blockIdx.z;
  const int tid = threadIdx.x, w = tid >> 5, lane = tid & 31;

  const bf16* qg = Qb + ((size_t)h * 8192 + (size_t)b * S + q0) * D;
  const bf16* kg = Kb + ((size_t)h * 8192 + (size_t)b * S) * D;
  const bf16* vg = Vt + (((size_t)h * 4 + b) * 512) * (size_t)2048;

  // Stage Q block into LDS (reused across the whole K loop).
#if USE_TDM
  if (w == 0)
    tdm_load_2d(QS_OFF, qg, 512, 64, 512, /*pad 256+4 dw*/ 7, 3, true);
#else
  for (int c = tid; c < 4096; c += 128) {
    int r = c >> 6, ko = (c & 63) * 8;
    *(v8bf*)&qs[r * 520 + ko] = *(const v8bf*)&qg[(size_t)r * D + ko];
  }
#endif

  v8f acc[32];
#pragma unroll
  for (int t = 0; t < 32; ++t)
#pragma unroll
    for (int j = 0; j < 8; ++j) acc[t][j] = 0.0f;
  float m_i[8], l_i[8];
#pragma unroll
  for (int j = 0; j < 8; ++j) { m_i[j] = -3.0e38f; l_i[j] = 0.0f; }

  const int qw = q0 + w * 16;
  const int nkt = (q0 + 63) / 32 + 1;
  bf16* pw = ps + w * 16 * 40;

  for (int kt = 0; kt < nkt; ++kt) {
    __syncthreads();  // previous tile fully consumed
#if USE_TDM
    if (w == 0) {
      tdm_load_2d(KS_OFF, kg + (size_t)(kt * 32) * D, 512, 32, 512, 7, 3, true);
      tdm_load_2d(VS_OFF, vg + kt * 32, 32, 512, 2048, /*pad 16+4 dw*/ 3, 3, true);
      __builtin_amdgcn_s_wait_tensorcnt(0);
    }
#else
    for (int c = tid; c < 2048; c += 128) {
      int r = c >> 6, ko = (c & 63) * 8;
      *(v8bf*)&ks[r * 520 + ko] =
          *(const v8bf*)&kg[(size_t)(kt * 32 + r) * D + ko];
    }
    for (int c = tid; c < 2048; c += 128) {
      int r = c >> 2, ko = (c & 3) * 8;
      *(v8bf*)&vs[r * 40 + ko] =
          *(const v8bf*)&vg[(size_t)r * 2048 + kt * 32 + ko];
    }
#endif
    __syncthreads();

    // S = Q * K^T  (two 16x16 C tiles covering 32 keys), K-depth 512.
    v8f s0, s1;
#pragma unroll
    for (int j = 0; j < 8; ++j) { s0[j] = 0.0f; s1[j] = 0.0f; }
#pragma unroll
    for (int kc = 0; kc < 16; ++kc) {
      v16bf a  = load_A_frag(&qs[w * 16 * 520 + kc * 32], 520);
      v16bf b0 = load_B_frag(&ks[kc * 32], 520);
      v16bf b1 = load_B_frag(&ks[16 * 520 + kc * 32], 520);
      s0 = wmma_bf16(a, b0, s0);
      s1 = wmma_bf16(a, b1, s1);
    }

    // Online softmax (row stats live per C-tile row: VGPR j <-> row).
    const int rowbase = qw + (lane >> 4) * 8;
    const int col0 = kt * 32 + (lane & 15);
    float alpha[8];
#pragma unroll
    for (int j = 0; j < 8; ++j) {
      int row = rowbase + j;
      float x0 = s0[j] * 0.125f;   // / sqrt(d_model/num_heads) = /8
      float x1 = s1[j] * 0.125f;
      if (col0 > row)      x0 = -3.0e38f;
      if (col0 + 16 > row) x1 = -3.0e38f;
      float mx = fmaxf(x0, x1);
      mx = fmaxf(mx, __shfl_xor(mx, 1));
      mx = fmaxf(mx, __shfl_xor(mx, 2));
      mx = fmaxf(mx, __shfl_xor(mx, 4));
      mx = fmaxf(mx, __shfl_xor(mx, 8));
      float mnew = fmaxf(m_i[j], mx);
      float e0 = __expf(x0 - mnew);
      float e1 = __expf(x1 - mnew);
      float rs = e0 + e1;
      rs += __shfl_xor(rs, 1);
      rs += __shfl_xor(rs, 2);
      rs += __shfl_xor(rs, 4);
      rs += __shfl_xor(rs, 8);
      alpha[j] = __expf(m_i[j] - mnew);
      l_i[j] = l_i[j] * alpha[j] + rs;
      m_i[j] = mnew;
      int pr = j + (lane >> 4) * 8;
      pw[pr * 40 + (lane & 15)]      = (bf16)e0;
      pw[pr * 40 + 16 + (lane & 15)] = (bf16)e1;
    }
#pragma unroll
    for (int t = 0; t < 32; ++t)
#pragma unroll
      for (int j = 0; j < 8; ++j) acc[t][j] *= alpha[j];

    asm volatile("s_wait_dscnt 0" ::: "memory");  // same-wave LDS RAW fence
    v16bf pa = load_A_frag(pw, 40);
#pragma unroll
    for (int t = 0; t < 32; ++t) {
      v16bf bvf = load_B_frag(&vs[t * 16 * 40], 40);
      acc[t] = wmma_bf16(pa, bvf, acc[t]);
    }
  }

  // Normalize and scatter into cat[B*S][H*512] (bf16).
#pragma unroll
  for (int j = 0; j < 8; ++j) l_i[j] = 1.0f / l_i[j];
  const int gbase = b * S + qw;
#pragma unroll
  for (int t = 0; t < 32; ++t) {
    int col = h * 512 + t * 16 + (lane & 15);
#pragma unroll
    for (int j = 0; j < 8; ++j) {
      int g = gbase + j + (lane >> 4) * 8;
      cat[(size_t)g * 4096 + col] = (bf16)(acc[t][j] * l_i[j]);
    }
  }
}

// ---------------------------------------------------------------------------
// Kernel 4: output projection  out[8192][512] = cat[8192][4096] @ Wp + bp
// WG tile 128x64, 8 WMMA / wave / K-step.
// ---------------------------------------------------------------------------
__global__ __launch_bounds__(128) void k_out_gemm(
    const bf16* __restrict__ catb, const bf16* __restrict__ wpT,
    const float* __restrict__ bp, float* __restrict__ out) {
  const int K = 4096, N = 512;
  __shared__ __align__(16) bf16 As[128 * 40];
  __shared__ __align__(16) bf16 Bs[64 * 40];
  const int m0 = blockIdx.x * 128;
  const int n0 = blockIdx.y * 64;
  const int tid = threadIdx.x, w = tid >> 5, lane = tid & 31;

  v8f acc[2][4];
#pragma unroll
  for (int mi = 0; mi < 2; ++mi)
#pragma unroll
    for (int t = 0; t < 4; ++t)
#pragma unroll
      for (int j = 0; j < 8; ++j) acc[mi][t][j] = 0.0f;

  for (int kk = 0; kk < K; kk += 32) {
#pragma unroll
    for (int it = 0; it < 4; ++it) {
      int c = tid + it * 128;
      int r = c >> 2, ko = (c & 3) * 8;
      *(v8bf*)&As[r * 40 + ko] =
          *(const v8bf*)&catb[(size_t)(m0 + r) * K + kk + ko];
    }
#pragma unroll
    for (int it = 0; it < 2; ++it) {
      int c = tid + it * 128;
      int r = c >> 2, ko = (c & 3) * 8;
      *(v8bf*)&Bs[r * 40 + ko] =
          *(const v8bf*)&wpT[(size_t)(n0 + r) * K + kk + ko];
    }
    __syncthreads();
    v16bf a0 = load_A_frag(&As[(w * 32) * 40], 40);
    v16bf a1 = load_A_frag(&As[(w * 32 + 16) * 40], 40);
#pragma unroll
    for (int t = 0; t < 4; ++t) {
      v16bf b = load_B_frag(&Bs[t * 16 * 40], 40);
      acc[0][t] = wmma_bf16(a0, b, acc[0][t]);
      acc[1][t] = wmma_bf16(a1, b, acc[1][t]);
    }
    __syncthreads();
  }

#pragma unroll
  for (int mi = 0; mi < 2; ++mi)
#pragma unroll
    for (int t = 0; t < 4; ++t) {
      int col = n0 + t * 16 + (lane & 15);
      float bval = bp[col];
#pragma unroll
      for (int j = 0; j < 8; ++j) {
        int row = m0 + w * 32 + mi * 16 + j + (lane >> 4) * 8;
        out[(size_t)row * N + col] = acc[mi][t][j] + bval;
      }
    }
}

// ---------------------------------------------------------------------------
// Host launcher
// ---------------------------------------------------------------------------
extern "C" void kernel_launch(void* const* d_in, const int* in_sizes, int n_in,
                              void* d_out, int out_size, void* d_ws, size_t ws_size,
                              hipStream_t stream) {
  const float* x  = (const float*)d_in[0];
  const float* Wq = (const float*)d_in[1];
  const float* bq = (const float*)d_in[2];
  const float* Wk = (const float*)d_in[3];
  const float* bk = (const float*)d_in[4];
  const float* Wv = (const float*)d_in[5];
  const float* bv = (const float*)d_in[6];
  const float* Wp = (const float*)d_in[7];
  const float* bp = (const float*)d_in[8];
  float* out = (float*)d_out;

  char* p = (char*)d_ws;
  bf16* xb  = (bf16*)p; p += (size_t)8192 * 512 * 2;
  bf16* wqT = (bf16*)p; p += (size_t)8 * 512 * 512 * 2;
  bf16* wkT = (bf16*)p; p += (size_t)8 * 512 * 512 * 2;
  bf16* wvT = (bf16*)p; p += (size_t)8 * 512 * 512 * 2;
  bf16* wpT = (bf16*)p; p += (size_t)512 * 4096 * 2;
  bf16* Qb  = (bf16*)p; p += (size_t)8 * 8192 * 512 * 2;
  bf16* Kb  = (bf16*)p; p += (size_t)8 * 8192 * 512 * 2;
  bf16* Vt  = (bf16*)p; p += (size_t)8 * 4 * 512 * 2048 * 2;
  bf16* cat = (bf16*)p; p += (size_t)8192 * 4096 * 2;

  int nX = 8192 * 512;
  k_f32_to_bf16<<<(nX + 255) / 256, 256, 0, stream>>>(x, xb, nX);
  int nW = 8 * 512 * 512;
  k_transpose_w<<<(nW + 255) / 256, 256, 0, stream>>>(Wq, wqT, 8, 512, 512);
  k_transpose_w<<<(nW + 255) / 256, 256, 0, stream>>>(Wk, wkT, 8, 512, 512);
  k_transpose_w<<<(nW + 255) / 256, 256, 0, stream>>>(Wv, wvT, 8, 512, 512);
  int nP = 4096 * 512;
  k_transpose_w<<<(nP + 255) / 256, 256, 0, stream>>>(Wp, wpT, 1, 4096, 512);

  k_qkv_gemm<<<dim3(64, 8, 24), 128, 0, stream>>>(xb, wqT, wkT, wvT, bq, bk, bv,
                                                  Qb, Kb, Vt);

  size_t flash_lds =
      (size_t)(64 * 520 + 32 * 520 + 512 * 40 + 4 * 16 * 40) * sizeof(bf16);
  k_flash<<<dim3(32, 4, 8), 128, flash_lds, stream>>>(Qb, Kb, Vt, cat);

  k_out_gemm<<<dim3(64, 8), 128, 0, stream>>>(cat, wpT, bp, out);
}
